// AdaOctConv_72997264163385
// MI455X (gfx1250) — compile-verified, matrix operand order checked by reading, not run on hardware
//
#include <hip/hip_runtime.h>

#define BDIM 4
#define CCH  128
#define SCH  512
#define HHF  256
#define HLF  128
#define EPSV 1e-5f

#define MT   4          // M tiles (of 16 pixels) per wave -> 64 px per block
#define COLS (16 * MT + 2)

typedef __attribute__((ext_vector_type(16))) _Float16 v16h;
typedef __attribute__((ext_vector_type(8)))  _Float16 v8h;
typedef __attribute__((ext_vector_type(8)))  float    v8f;

__device__ __forceinline__ int refl(int p, int n) {
    p = p < 0 ? -p : p;
    return p >= n ? 2 * n - 2 - p : p;
}

// ---------------------------------------------------------------------------
// Kernel predictor + fusion of pointwise into spatial weights.
// One block per batch sample, 256 threads.
// Produces fusedW[b][c][d'][3][3] (18 floats/channel) and bias[b][c].
// ---------------------------------------------------------------------------
__global__ void predict_fuse_kernel(const float* __restrict__ s,
                                    const float* __restrict__ sp_w, const float* __restrict__ sp_b,
                                    const float* __restrict__ pw_w, const float* __restrict__ pw_b,
                                    const float* __restrict__ bw,   const float* __restrict__ bb,
                                    float* __restrict__ fusedW, float* __restrict__ biasOut)
{
    __shared__ float s_lds[SCH * 9];
    __shared__ float sm_lds[SCH];
    __shared__ float sp_lds[256 * 9];
    __shared__ float wp_lds[256];
    const int b = blockIdx.x;
    const int tid = threadIdx.x;

    for (int i = tid; i < SCH * 9; i += 256) s_lds[i] = s[b * SCH * 9 + i];
    __syncthreads();

    for (int c = tid; c < SCH; c += 256) {
        float m = 0.f;
        #pragma unroll
        for (int p = 0; p < 9; ++p) m += s_lds[c * 9 + p];
        sm_lds[c] = m * (1.f / 9.f);
    }
    __syncthreads();

    // spatial predictor: reflect-pad(1) 3x3 conv, 3x3 input -> 3x3 output
    {
        const int oc = tid;                 // 256 out channels
        const int mp[5] = {1, 0, 1, 2, 1};  // reflect index map for padded 5
        float acc[9];
        #pragma unroll
        for (int p = 0; p < 9; ++p) acc[p] = sp_b[oc];
        for (int ic = 0; ic < SCH; ++ic) {
            float sv[9];
            #pragma unroll
            for (int p = 0; p < 9; ++p) sv[p] = s_lds[ic * 9 + p];
            #pragma unroll
            for (int ku = 0; ku < 3; ++ku)
            #pragma unroll
            for (int kv = 0; kv < 3; ++kv) {
                float w = sp_w[((oc * SCH + ic) * 3 + ku) * 3 + kv];
                #pragma unroll
                for (int oy = 0; oy < 3; ++oy)
                #pragma unroll
                for (int ox = 0; ox < 3; ++ox)
                    acc[oy * 3 + ox] = fmaf(w, sv[mp[oy + ku] * 3 + mp[ox + kv]], acc[oy * 3 + ox]);
            }
        }
        #pragma unroll
        for (int p = 0; p < 9; ++p) sp_lds[oc * 9 + p] = acc[p];
    }
    // pointwise predictor (matvec over style mean)
    {
        float acc = pw_b[tid];
        for (int ic = 0; ic < SCH; ++ic) acc = fmaf(sm_lds[ic], pw_w[tid * SCH + ic], acc);
        wp_lds[tid] = acc;
    }
    if (tid < CCH) {
        float acc = bb[tid];
        for (int ic = 0; ic < SCH; ++ic) acc = fmaf(sm_lds[ic], bw[tid * SCH + ic], acc);
        biasOut[b * CCH + tid] = acc;
    }
    __syncthreads();

    // fuse: W_fused[c,d'] = wp[c,0]*ws[2g+0,d'] + wp[c,1]*ws[2g+1,d']  (g = c/2)
    for (int idx = tid; idx < CCH * 18; idx += 256) {
        int c = idx / 18, r = idx % 18, dp = r / 9, uv = r % 9;
        int g2 = c & ~1;
        float f = wp_lds[c * 2 + 0] * sp_lds[((g2    ) * 2 + dp) * 9 + uv]
                + wp_lds[c * 2 + 1] * sp_lds[((g2 + 1) * 2 + dp) * 9 + uv];
        fusedW[(b * CCH + c) * 18 + dp * 9 + uv] = f;
    }
}

// ---------------------------------------------------------------------------
// Per-(b,c) mean / rstd over spatial (instance-norm statistics).
// ---------------------------------------------------------------------------
__global__ void stats_kernel(const float* __restrict__ x, int hw, float* __restrict__ stats)
{
    __shared__ float rs[256], rq[256];
    const int bc = blockIdx.x;
    const float* p = x + (size_t)bc * hw;
    float s0 = 0.f, s1 = 0.f;
    for (int i = threadIdx.x; i < hw; i += 256) { float v = p[i]; s0 += v; s1 = fmaf(v, v, s1); }
    rs[threadIdx.x] = s0; rq[threadIdx.x] = s1;
    __syncthreads();
    for (int o = 128; o > 0; o >>= 1) {
        if (threadIdx.x < o) { rs[threadIdx.x] += rs[threadIdx.x + o]; rq[threadIdx.x] += rq[threadIdx.x + o]; }
        __syncthreads();
    }
    if (threadIdx.x == 0) {
        float m = rs[0] / hw;
        float v = rq[0] / hw - m * m;
        stats[2 * bc]     = m;
        stats[2 * bc + 1] = rsqrtf(v + EPSV);
    }
}

// ---------------------------------------------------------------------------
// Fused: instance-norm(x) -> grouped spatial+pointwise conv (fused weights)
// + bias, write fp16 mid activation in NHWC ([b][i][j][cin], cin innermost).
// Threads = (channel, half-row): channel-contiguous 2B stores coalesce to
// 256 B runs; NHWC makes the WMMA kernel's LDS staging a pure block copy.
// ---------------------------------------------------------------------------
__global__ void mid_kernel(const float* __restrict__ x, const float* __restrict__ stats,
                           const float* __restrict__ fusedW, const float* __restrict__ biasArr,
                           _Float16* __restrict__ yt, int Hd)
{
    const int i = blockIdx.x, b = blockIdx.y;
    const int c = threadIdx.x & (CCH - 1);
    const int half = threadIdx.x >> 7;
    const int cin0 = c & ~1;

    const float* fwp = fusedW + (b * CCH + c) * 18;
    float w[18];
    #pragma unroll
    for (int p = 0; p < 18; ++p) w[p] = fwp[p];
    const float bias = biasArr[b * CCH + c];

    float mean[2], rstd[2];
    const float* xp[2];
    #pragma unroll
    for (int dp = 0; dp < 2; ++dp) {
        mean[dp] = stats[2 * (b * CCH + cin0 + dp)];
        rstd[dp] = stats[2 * (b * CCH + cin0 + dp) + 1];
        xp[dp]   = x + (size_t)(b * CCH + cin0 + dp) * Hd * Hd;
    }
    const int ro[3] = { refl(i - 1, Hd), i, refl(i + 1, Hd) };

    const int jb = half * (Hd >> 1), je = jb + (Hd >> 1);
    for (int j = jb; j < je; ++j) {
        float acc = bias;
        #pragma unroll
        for (int dp = 0; dp < 2; ++dp)
        #pragma unroll
        for (int u = 0; u < 3; ++u) {
            const float* row = xp[dp] + (size_t)ro[u] * Hd;
            #pragma unroll
            for (int v = 0; v < 3; ++v) {
                int cj = refl(j - 1 + v, Hd);
                acc = fmaf(w[dp * 9 + u * 3 + v], (row[cj] - mean[dp]) * rstd[dp], acc);
            }
        }
        yt[(((size_t)b * Hd + i) * Hd + j) * CCH + c] = (_Float16)acc;
    }
}

// ---------------------------------------------------------------------------
// Transpose+convert shared conv weights: [cout][cin][3][3] f32
//   -> wT[cout][tap][cin] f16 (cin innermost -> contiguous B-fragment loads)
// ---------------------------------------------------------------------------
__global__ void wconv_kernel(const float* __restrict__ w, _Float16* __restrict__ wT)
{
    int idx = blockIdx.x * 256 + threadIdx.x;
    if (idx >= CCH * CCH * 9) return;
    int cout = idx / (CCH * 9);
    int rem  = idx % (CCH * 9);
    int cin  = rem / 9;
    int uv   = rem % 9;
    wT[(cout * 9 + uv) * CCH + cin] = (_Float16)w[idx];
}

// ---------------------------------------------------------------------------
// Dense 3x3 conv 128->128 as implicit GEMM with V_WMMA_F32_16X16X32_F16.
// Workgroup = 8 waves: 64 output pixels x 128 output channels.
//  - NHWC input => staging = contiguous 16B block copies, issued with
//    GLOBAL_LOAD_ASYNC_TO_LDS_B128 (ASYNCcnt), drained by s_wait_asynccnt.
//  - M-register-blocked: B (weight) fragment reused across MT=4 A fragments.
//  - A fragments batched into an array before the 4 WMMAs so ds_load_b128
//    issue in one clause instead of lockstep load/wait/wmma.
// 9 taps x 4 k-chunks x 4 M-tiles = 144 WMMA per wave.
// ---------------------------------------------------------------------------
__global__ void conv3x3_wmma_kernel(const _Float16* __restrict__ y, const _Float16* __restrict__ wT,
                                    const float* __restrict__ cb, float* __restrict__ z, int Hd)
{
    __shared__ _Float16 tile[3 * COLS * CCH];  // 3*66*128*2B = 50688 B
    const int j0 = blockIdx.x * (16 * MT);
    const int i  = blockIdx.y;
    const int b  = blockIdx.z;
    const int tid = threadIdx.x;

    // stage reflect-padded patch rows (di,jj): each is 128 cin = 256 B contiguous
    const int nchunk = 3 * COLS * (CCH / 8);   // 16-byte chunks
    for (int t = tid; t < nchunk; t += 256) {
        const int row = t >> 4;                // (di, jj)
        const int q   = t & 15;                // 16B chunk within the 256B row
        const int di = row / COLS, jj = row % COLS;
        const int ri = refl(i - 1 + di, Hd);
        const int cj = refl(j0 - 1 + jj, Hd);
        const _Float16* src = y + (((size_t)b * Hd + ri) * Hd + cj) * CCH + q * 8;
        const _Float16* dst = tile + (di * COLS + jj) * CCH + q * 8;
        unsigned ldsoff = (unsigned)(uintptr_t)dst;
        asm volatile("global_load_async_to_lds_b128 %0, %1, off"
                     :: "v"(ldsoff), "v"(src) : "memory");
    }
    asm volatile("s_wait_asynccnt 0" ::: "memory");
    __syncthreads();

    const int wave  = tid >> 5;
    const int lane  = tid & 31;
    const int hi    = lane >> 4;
    const int laneN = lane & 15;
    const int ch    = wave * 16 + laneN;   // N index = out channel

    const _Float16* wrow = wT + (size_t)ch * (9 * CCH);
    __builtin_prefetch(wrow, 0, 0);        // global_prefetch: this channel's 2.3 KB weight slab

    v8f acc[MT] = {};
    #pragma unroll
    for (int uv = 0; uv < 9; ++uv) {
        const int u = uv / 3, v = uv % 3;
        const _Float16* wb = wrow + uv * CCH;
        #pragma unroll
        for (int kc = 0; kc < 4; ++kc) {
            const int kb = kc * 32 + hi * 8;
            v8h b0 = *(const v8h*)(wb + kb);
            v8h b1 = *(const v8h*)(wb + kb + 16);
            v16h bm = __builtin_shufflevector(b0, b1, 0,1,2,3,4,5,6,7,8,9,10,11,12,13,14,15);
            v16h afr[MT];
            #pragma unroll
            for (int m = 0; m < MT; ++m) {
                // A: M = lane&15 (+16*m spatial col), K halves kb..kb+7, kb+16..kb+23
                const _Float16* ab = &tile[((u * COLS) + laneN + v + 16 * m) * CCH];
                v8h a0 = *(const v8h*)(ab + kb);
                v8h a1 = *(const v8h*)(ab + kb + 16);
                afr[m] = __builtin_shufflevector(a0, a1, 0,1,2,3,4,5,6,7,8,9,10,11,12,13,14,15);
            }
            #pragma unroll
            for (int m = 0; m < MT; ++m)
                acc[m] = __builtin_amdgcn_wmma_f32_16x16x32_f16(
                             false, afr[m], false, bm, (short)0, acc[m], false, false);
        }
    }

    // C/D layout: lane -> N (out channel), VGPR r -> M = r + 8*hi (spatial col)
    const float bias = cb[ch];
    float* zp = z + ((size_t)(b * CCH + ch) * Hd + i) * Hd + j0 + 8 * hi;
    #pragma unroll
    for (int m = 0; m < MT; ++m)
        #pragma unroll
        for (int r = 0; r < 8; ++r) zp[16 * m + r] = acc[m][r] + bias;
}

// ---------------------------------------------------------------------------
// In-place instance-norm + LeakyReLU(0.2) on the conv output.
// ---------------------------------------------------------------------------
__global__ void finalize_kernel(float* __restrict__ z, const float* __restrict__ stats,
                                int hw, int total)
{
    for (int idx = blockIdx.x * 256 + threadIdx.x; idx < total; idx += gridDim.x * 256) {
        int bc = idx / hw;
        float m = stats[2 * bc], r = stats[2 * bc + 1];
        float v = (z[idx] - m) * r;
        z[idx] = v >= 0.f ? v : 0.2f * v;
    }
}

extern "C" void kernel_launch(void* const* d_in, const int* in_sizes, int n_in,
                              void* d_out, int out_size, void* d_ws, size_t ws_size,
                              hipStream_t stream)
{
    const float* c_hf      = (const float*)d_in[0];
    const float* c_lf      = (const float*)d_in[1];
    const float* s_hf      = (const float*)d_in[2];
    const float* s_lf      = (const float*)d_in[3];
    const float* kp_h_sp_w = (const float*)d_in[4];
    const float* kp_h_sp_b = (const float*)d_in[5];
    const float* kp_h_pw_w = (const float*)d_in[6];
    const float* kp_h_pw_b = (const float*)d_in[7];
    const float* kp_h_b_w  = (const float*)d_in[8];
    const float* kp_h_b_b  = (const float*)d_in[9];
    const float* kp_l_sp_w = (const float*)d_in[10];
    const float* kp_l_sp_b = (const float*)d_in[11];
    const float* kp_l_pw_w = (const float*)d_in[12];
    const float* kp_l_pw_b = (const float*)d_in[13];
    const float* kp_l_b_w  = (const float*)d_in[14];
    const float* kp_l_b_b  = (const float*)d_in[15];
    const float* conv_h_w  = (const float*)d_in[16];
    const float* conv_h_b  = (const float*)d_in[17];
    const float* conv_l_w  = (const float*)d_in[18];
    const float* conv_l_b  = (const float*)d_in[19];

    float* out = (float*)d_out;
    char* ws = (char*)d_ws;
    size_t off = 0;
    auto alloc = [&](size_t bytes) -> char* {
        char* p = ws + off;
        off = (off + bytes + 255) & ~(size_t)255;
        return p;
    };
    float*    fw_h = (float*)alloc((size_t)BDIM * CCH * 18 * 4);
    float*    bs_h = (float*)alloc((size_t)BDIM * CCH * 4);
    float*    fw_l = (float*)alloc((size_t)BDIM * CCH * 18 * 4);
    float*    bs_l = (float*)alloc((size_t)BDIM * CCH * 4);
    float*    xs_h = (float*)alloc((size_t)BDIM * CCH * 2 * 4);
    float*    xs_l = (float*)alloc((size_t)BDIM * CCH * 2 * 4);
    float*    zs_h = (float*)alloc((size_t)BDIM * CCH * 2 * 4);
    float*    zs_l = (float*)alloc((size_t)BDIM * CCH * 2 * 4);
    _Float16* wT_h = (_Float16*)alloc((size_t)9 * CCH * CCH * 2);
    _Float16* wT_l = (_Float16*)alloc((size_t)9 * CCH * CCH * 2);
    _Float16* y_h  = (_Float16*)alloc((size_t)BDIM * CCH * HHF * HHF * 2);
    _Float16* y_l  = (_Float16*)alloc((size_t)BDIM * CCH * HLF * HLF * 2);

    float* z_h = out;
    float* z_l = out + (size_t)BDIM * CCH * HHF * HHF;

    predict_fuse_kernel<<<BDIM, 256, 0, stream>>>(
        s_hf, kp_h_sp_w, kp_h_sp_b, kp_h_pw_w, kp_h_pw_b, kp_h_b_w, kp_h_b_b, fw_h, bs_h);
    predict_fuse_kernel<<<BDIM, 256, 0, stream>>>(
        s_lf, kp_l_sp_w, kp_l_sp_b, kp_l_pw_w, kp_l_pw_b, kp_l_b_w, kp_l_b_b, fw_l, bs_l);

    stats_kernel<<<BDIM * CCH, 256, 0, stream>>>(c_hf, HHF * HHF, xs_h);
    stats_kernel<<<BDIM * CCH, 256, 0, stream>>>(c_lf, HLF * HLF, xs_l);

    mid_kernel<<<dim3(HHF, BDIM), 256, 0, stream>>>(c_hf, xs_h, fw_h, bs_h, y_h, HHF);
    mid_kernel<<<dim3(HLF, BDIM), 256, 0, stream>>>(c_lf, xs_l, fw_l, bs_l, y_l, HLF);

    wconv_kernel<<<(CCH * CCH * 9 + 255) / 256, 256, 0, stream>>>(conv_h_w, wT_h);
    wconv_kernel<<<(CCH * CCH * 9 + 255) / 256, 256, 0, stream>>>(conv_l_w, wT_l);

    conv3x3_wmma_kernel<<<dim3(HHF / (16 * MT), HHF, BDIM), 256, 0, stream>>>(y_h, wT_h, conv_h_b, z_h, HHF);
    conv3x3_wmma_kernel<<<dim3(HLF / (16 * MT), HLF, BDIM), 256, 0, stream>>>(y_l, wT_l, conv_l_b, z_l, HLF);

    stats_kernel<<<BDIM * CCH, 256, 0, stream>>>(z_h, HHF * HHF, zs_h);
    stats_kernel<<<BDIM * CCH, 256, 0, stream>>>(z_l, HLF * HLF, zs_l);

    finalize_kernel<<<4096, 256, 0, stream>>>(z_h, zs_h, HHF * HHF, BDIM * CCH * HHF * HHF);
    finalize_kernel<<<1024, 256, 0, stream>>>(z_l, zs_l, HLF * HLF, BDIM * CCH * HLF * HLF);
}
